// BinaryGramLanguageModel_54357106098774
// MI455X (gfx1250) — matrix operand order repaired and checked by reading, not compiled, not run on hardware
//
#include <hip/hip_runtime.h>
#include <hip/hip_bf16.h>

// ---------------------------------------------------------------------------
// GPT forward pass for MI455X (gfx1250, wave32, WMMA).
// L=6, E=1024, H=16, HD=64, T=1024, B=4, V=32000.
// ~1 TFLOP total, weights ~550MB fp32 (~25us @ 23.3TB/s) => compute bound:
// all GEMMs and attention run on v_wmma_f32_16x16x32_f16.
// f16 activation tiles are staged with global_load_async_to_lds_b128
// (ASYNCcnt), f32 weights are converted f32->f16 in registers and stored
// packed (b32) into a transposed, bank-conflict-padded LDS tile.
// ---------------------------------------------------------------------------

typedef _Float16 v16h __attribute__((ext_vector_type(16)));
typedef _Float16 v8h  __attribute__((ext_vector_type(8)));
typedef _Float16 v4h  __attribute__((ext_vector_type(4)));
typedef float    v8f  __attribute__((ext_vector_type(8)));

#define DEVINL static __device__ __forceinline__

DEVINL v16h ldfrag(const _Float16* lo, const _Float16* hi) {
    v8h a = *reinterpret_cast<const v8h*>(lo);
    v8h b = *reinterpret_cast<const v8h*>(hi);
    return __builtin_shufflevector(a, b, 0,1,2,3,4,5,6,7,8,9,10,11,12,13,14,15);
}

DEVINL v8f wmma16(v16h a, v16h b, v8f c) {
    // D = A(16x32 f16) x B(32x16 f16) + C(16x16 f32)
    return __builtin_amdgcn_wmma_f32_16x16x32_f16(
        /*neg_a=*/false, a, /*neg_b=*/false, b,
        /*c_mod=*/(short)0, c, /*reuse_a=*/false, /*reuse_b=*/false);
}

// ---- CDNA5 async global->LDS copy (ASYNCcnt tracked) ----------------------
// VDST VGPR = per-lane LDS byte offset; generic pointers to __shared__ carry
// the LDS offset in the low 32 bits.
DEVINL void async_b128(uint32_t lds_off, const void* gaddr) {
    asm volatile("global_load_async_to_lds_b128 %0, %1, off"
                 :: "v"(lds_off), "v"(gaddr) : "memory");
}
DEVINL void wait_async0() {
    asm volatile("s_wait_asynccnt 0x0" ::: "memory");
}
DEVINL uint32_t lds_off(const void* p) { return (uint32_t)(uintptr_t)p; }

// ------------------------------ constants ----------------------------------
static constexpr int Lh  = 6;
static constexpr int Eh  = 1024;
static constexpr int Hh  = 16;
static constexpr int HDh = 64;
static constexpr int Th  = 1024;
static constexpr int Bh  = 4;
static constexpr int Vh  = 32000;
static constexpr int Mh  = Bh * Th;      // 4096 rows of activations
static constexpr int PAD = 8;            // LDS row pad (f16) -> 144B stride

// ------------------------------ embedding ----------------------------------
__global__ __launch_bounds__(256) void embed_kernel(
    const int* __restrict__ idx, const float* __restrict__ tok,
    const float* __restrict__ pos, float* __restrict__ x)
{
    int row = blockIdx.x;                 // 0..4095  (b*T + t)
    int t   = row & (Th - 1);
    int tokid = idx[row];
    int c = threadIdx.x;                  // 256 threads * float4 = 1024 floats
    float4 a = reinterpret_cast<const float4*>(tok + (size_t)tokid * Eh)[c];
    float4 p = reinterpret_cast<const float4*>(pos + (size_t)t * Eh)[c];
    float4 o; o.x = a.x + p.x; o.y = a.y + p.y; o.z = a.z + p.z; o.w = a.w + p.w;
    reinterpret_cast<float4*>(x + (size_t)row * Eh)[c] = o;
}

// ------------------------------ layernorm ----------------------------------
__global__ __launch_bounds__(128) void layernorm_kernel(
    const float* __restrict__ x, const float* __restrict__ g,
    const float* __restrict__ b, _Float16* __restrict__ out)
{
    int row = blockIdx.x;
    int tid = threadIdx.x, lane = tid & 31, wid = tid >> 5;
    const float* xr = x + (size_t)row * Eh;
    float4 v1 = reinterpret_cast<const float4*>(xr)[tid];
    float4 v2 = reinterpret_cast<const float4*>(xr)[tid + 128];
    float s  = v1.x + v1.y + v1.z + v1.w + v2.x + v2.y + v2.z + v2.w;
    float ss = v1.x*v1.x + v1.y*v1.y + v1.z*v1.z + v1.w*v1.w
             + v2.x*v2.x + v2.y*v2.y + v2.z*v2.z + v2.w*v2.w;
    for (int m = 1; m < 32; m <<= 1) {
        s  += __shfl_xor(s,  m, 32);
        ss += __shfl_xor(ss, m, 32);
    }
    __shared__ float red[8];
    if (lane == 0) { red[wid] = s; red[wid + 4] = ss; }
    __syncthreads();
    s  = red[0] + red[1] + red[2] + red[3];
    ss = red[4] + red[5] + red[6] + red[7];
    float mean = s * (1.0f / Eh);
    float var  = ss * (1.0f / Eh) - mean * mean;
    float rstd = rsqrtf(var + 1e-5f);

    float4 g1 = reinterpret_cast<const float4*>(g)[tid];
    float4 b1 = reinterpret_cast<const float4*>(b)[tid];
    float4 g2 = reinterpret_cast<const float4*>(g)[tid + 128];
    float4 b2 = reinterpret_cast<const float4*>(b)[tid + 128];
    v4h o1, o2;
    o1[0] = (_Float16)((v1.x - mean) * rstd * g1.x + b1.x);
    o1[1] = (_Float16)((v1.y - mean) * rstd * g1.y + b1.y);
    o1[2] = (_Float16)((v1.z - mean) * rstd * g1.z + b1.z);
    o1[3] = (_Float16)((v1.w - mean) * rstd * g1.w + b1.w);
    o2[0] = (_Float16)((v2.x - mean) * rstd * g2.x + b2.x);
    o2[1] = (_Float16)((v2.y - mean) * rstd * g2.y + b2.y);
    o2[2] = (_Float16)((v2.z - mean) * rstd * g2.z + b2.z);
    o2[3] = (_Float16)((v2.w - mean) * rstd * g2.w + b2.w);
    *reinterpret_cast<v4h*>(out + (size_t)row * Eh + tid * 4)         = o1;
    *reinterpret_cast<v4h*>(out + (size_t)row * Eh + (tid + 128) * 4) = o2;
}

// ------------------------------ WMMA GEMM ----------------------------------
// C[M,N] = A[M,K](f16) * B[K,N](f32 weights, converted to f16 in LDS)
// Tile 128x128x64, 256 threads = 8 waves (2 in M x 4 in N), wave = 64x32.
template <bool OUT16, bool OUT32, bool BIAS, bool RELU, bool RESID>
__global__ __launch_bounds__(256) void gemm_wmma_kernel(
    const _Float16* __restrict__ A, const float* __restrict__ Bw,
    const float* __restrict__ bias, const float* __restrict__ resid,
    float* __restrict__ C32, _Float16* __restrict__ C16,
    int Mdim, int Ndim, int Kdim)
{
    constexpr int BM = 128, BN = 128, BK = 64;
    constexpr int LS = BK + PAD;          // padded LDS row stride (f16)
    __shared__ _Float16 As[BM * LS];      // [m][k] row-major
    __shared__ _Float16 Bs[BN * LS];      // [n][k] (transposed)

    int tid = threadIdx.x, lane = tid & 31, wid = tid >> 5;
    int waveM = wid >> 2, waveN = wid & 3;
    int r16 = lane & 15, rhi = lane >> 4;
    int m0 = blockIdx.y * BM, n0 = blockIdx.x * BN;

    v8f acc[4][2] = {};

    int arow = tid >> 1, acol = (tid & 1) * 32;   // A stage: 32 f16 per thread
    int bn   = (tid & 31) * 4;                    // B stage: float4 columns
    int bkp  = (tid >> 5) * 2;                    // starting even k row

    for (int k0 = 0; k0 < Kdim; k0 += BK) {
        __syncthreads();
        // ---- stage A via async global->LDS (64B per thread) ----
        {
            const char* src = reinterpret_cast<const char*>(
                A + (size_t)(m0 + arow) * Kdim + k0 + acol);
            uint32_t dst = lds_off(As + arow * LS + acol);
            async_b128(dst,      src);
            async_b128(dst + 16, src + 16);
            async_b128(dst + 32, src + 32);
            async_b128(dst + 48, src + 48);
        }
        // ---- stage B: cvt f32->f16, pack 2 K-rows per b32 LDS store ----
        #pragma unroll
        for (int kk = 0; kk < 4; ++kk) {
            int kr = bkp + kk * 16;               // even row in [0,64)
            const float* p0 = Bw + (size_t)(k0 + kr) * Ndim + n0 + bn;
            float4 w0 = *reinterpret_cast<const float4*>(p0);
            float4 w1 = *reinterpret_cast<const float4*>(p0 + Ndim);
            float a0[4] = {w0.x, w0.y, w0.z, w0.w};
            float a1[4] = {w1.x, w1.y, w1.z, w1.w};
            #pragma unroll
            for (int e = 0; e < 4; ++e) {
                union { _Float16 h[2]; uint32_t u; } pk;
                pk.h[0] = (_Float16)a0[e];
                pk.h[1] = (_Float16)a1[e];
                *reinterpret_cast<uint32_t*>(&Bs[(bn + e) * LS + kr]) = pk.u;
            }
        }
        if (k0 + BK < Kdim)   // warm next weight tile (global_prefetch)
            __builtin_prefetch(Bw + (size_t)(k0 + BK) * Ndim + n0 + bn, 0, 1);
        wait_async0();
        __syncthreads();

        // ---- 16 WMMAs per tile per wave ----
        #pragma unroll
        for (int ks = 0; ks < 2; ++ks) {
            v16h af[4];
            #pragma unroll
            for (int i = 0; i < 4; ++i) {
                int row = waveM * 64 + i * 16 + r16;
                int kb  = ks * 32 + rhi * 8;
                af[i] = ldfrag(&As[row * LS + kb], &As[row * LS + kb + 16]);
            }
            #pragma unroll
            for (int j = 0; j < 2; ++j) {
                int n   = waveN * 32 + j * 16 + r16;
                int kb2 = ks * 32 + rhi * 8;
                v16h bf = ldfrag(&Bs[n * LS + kb2], &Bs[n * LS + kb2 + 16]);
                #pragma unroll
                for (int i = 0; i < 4; ++i)
                    acc[i][j] = wmma16(af[i], bf, acc[i][j]);
            }
        }
    }

    // ---- epilogue: C/D layout: lane r16 = col, rows = rhi*8 + e ----
    #pragma unroll
    for (int i = 0; i < 4; ++i)
        #pragma unroll
        for (int j = 0; j < 2; ++j) {
            int n = n0 + waveN * 32 + j * 16 + r16;
            #pragma unroll
            for (int e = 0; e < 8; ++e) {
                int m = m0 + waveM * 64 + i * 16 + rhi * 8 + e;
                float v = acc[i][j][e];
                if (BIAS)  v += bias[n];
                if (RESID) v += resid[(size_t)m * Ndim + n];
                if (RELU)  v = fmaxf(v, 0.0f);
                if (OUT32) C32[(size_t)m * Ndim + n] = v;
                if (OUT16) C16[(size_t)m * Ndim + n] = (_Float16)v;
            }
        }
}

// ---------------------------- flash attention ------------------------------
// Reference computes wei[t,s] = K[t]·Q[s] with causal mask s<=t, softmax over
// s, out = wei @ V.  => standard causal flash attention with "query" rows
// taken from the K buffer and "key" rows from the Q buffer.
// One block = (b,h) x 64 t-rows. 4 waves, each wave: 16 rows x 64 cols.
__global__ __launch_bounds__(128) void attn_kernel(
    const _Float16* __restrict__ q, const _Float16* __restrict__ k,
    const _Float16* __restrict__ v, _Float16* __restrict__ att)
{
    constexpr int LS = 64 + PAD;       // padded LDS row stride (f16)
    int bh = blockIdx.y, b = bh >> 4, h = bh & 15;
    int tblk = blockIdx.x;
    int tid = threadIdx.x, lane = tid & 31, wave = tid >> 5;
    int r16 = lane & 15, rhi = lane >> 4;

    __shared__ _Float16 Kt[64 * LS];   // [t][d]  (A for the score GEMM)
    __shared__ _Float16 Qs[64 * LS];   // [s][d]  (B: lane col=s, K-dim=d contiguous)
    __shared__ _Float16 Vt[64 * LS];   // [d][s]  (B: lane col=d, K-dim=s contiguous)
    __shared__ _Float16 Pp[64 * LS];   // [t][s]  softmaxed probs, f16

    const size_t headoff = ((size_t)b * Th) * Eh + h * HDh;
    int srow = tid >> 1, shalf = (tid & 1) * 32;

    // stage K-rows for this t-block (async, lives for the whole kernel)
    {
        const char* src = reinterpret_cast<const char*>(
            k + headoff + (size_t)(tblk * 64 + srow) * Eh + shalf);
        uint32_t dst = lds_off(Kt + srow * LS + shalf);
        async_b128(dst,      src);
        async_b128(dst + 16, src + 16);
        async_b128(dst + 32, src + 32);
        async_b128(dst + 48, src + 48);
    }

    float mrow[8], lrow[8];
    #pragma unroll
    for (int e = 0; e < 8; ++e) { mrow[e] = -__builtin_huge_valf(); lrow[e] = 0.0f; }
    v8f o_acc[4] = {};

    for (int sblk = 0; sblk <= tblk; ++sblk) {
        __syncthreads();   // everyone done with previous Qs/Vt
        {   // stage Q-rows (row-major, async)
            const char* src = reinterpret_cast<const char*>(
                q + headoff + (size_t)(sblk * 64 + srow) * Eh + shalf);
            uint32_t dst = lds_off(Qs + srow * LS + shalf);
            async_b128(dst,      src);
            async_b128(dst + 16, src + 16);
            async_b128(dst + 32, src + 32);
            async_b128(dst + 48, src + 48);
        }
        #pragma unroll
        for (int c = 0; c < 4; ++c) {   // stage V transposed [d][s]
            int ch = tid + c * 128;
            int s = ch & 63, d8 = (ch >> 6) * 8;
            v8h vv = *reinterpret_cast<const v8h*>(
                v + headoff + (size_t)(sblk * 64 + s) * Eh + d8);
            #pragma unroll
            for (int e = 0; e < 8; ++e) Vt[(d8 + e) * LS + s] = vv[e];
        }
        wait_async0();
        __syncthreads();

        // ---- S = Kt x Qs^T  (wave: 16 t-rows x 64 s-cols) ----
        v16h ka0, ka1;
        {
            int row = wave * 16 + r16, kb = rhi * 8;
            ka0 = ldfrag(&Kt[row * LS + kb],      &Kt[row * LS + kb + 16]);
            ka1 = ldfrag(&Kt[row * LS + kb + 32], &Kt[row * LS + kb + 48]);
        }
        v8f s_acc[4] = {};
        #pragma unroll
        for (int j = 0; j < 4; ++j) {
            int n = j * 16 + r16, kb2 = rhi * 8;
            v16h b0 = ldfrag(&Qs[n * LS + kb2],      &Qs[n * LS + kb2 + 16]);
            v16h b1 = ldfrag(&Qs[n * LS + 32 + kb2], &Qs[n * LS + 32 + kb2 + 16]);
            s_acc[j] = wmma16(ka0, b0, s_acc[j]);
            s_acc[j] = wmma16(ka1, b1, s_acc[j]);
        }

        // ---- scale, causal mask, online softmax ----
        float rmax[8];
        #pragma unroll
        for (int e = 0; e < 8; ++e) rmax[e] = -__builtin_huge_valf();
        #pragma unroll
        for (int j = 0; j < 4; ++j) {
            int sg = sblk * 64 + j * 16 + r16;
            #pragma unroll
            for (int e = 0; e < 8; ++e) {
                int tg = tblk * 64 + wave * 16 + rhi * 8 + e;
                float sv = s_acc[j][e] * 0.125f;             // HD^-0.5
                if (sg > tg) sv = -__builtin_huge_valf();
                s_acc[j][e] = sv;
                rmax[e] = fmaxf(rmax[e], sv);
            }
        }
        #pragma unroll
        for (int m = 1; m < 16; m <<= 1)   // reduce over 16-lane col group
            #pragma unroll
            for (int e = 0; e < 8; ++e)
                rmax[e] = fmaxf(rmax[e], __shfl_xor(rmax[e], m, 32));

        float fac[8], rsum[8];
        #pragma unroll
        for (int e = 0; e < 8; ++e) {
            float nm = fmaxf(mrow[e], rmax[e]);
            fac[e] = __expf(mrow[e] - nm);
            mrow[e] = nm;
            rsum[e] = 0.0f;
        }
        #pragma unroll
        for (int j = 0; j < 4; ++j)
            #pragma unroll
            for (int e = 0; e < 8; ++e) {
                float p = __expf(s_acc[j][e] - mrow[e]);
                rsum[e] += p;
                Pp[(wave * 16 + rhi * 8 + e) * LS + j * 16 + r16] = (_Float16)p;
            }
        #pragma unroll
        for (int m = 1; m < 16; m <<= 1)
            #pragma unroll
            for (int e = 0; e < 8; ++e)
                rsum[e] += __shfl_xor(rsum[e], m, 32);
        #pragma unroll
        for (int e = 0; e < 8; ++e) lrow[e] = lrow[e] * fac[e] + rsum[e];
        #pragma unroll
        for (int j = 0; j < 4; ++j)
            #pragma unroll
            for (int e = 0; e < 8; ++e) o_acc[j][e] *= fac[e];

        // ---- O += P @ V  (A = wave's own 16 P-rows, B = Vt) ----
        v16h pa0, pa1;
        {
            int row = wave * 16 + r16, kb = rhi * 8;
            pa0 = ldfrag(&Pp[row * LS + kb],      &Pp[row * LS + kb + 16]);
            pa1 = ldfrag(&Pp[row * LS + kb + 32], &Pp[row * LS + kb + 48]);
        }
        #pragma unroll
        for (int j = 0; j < 4; ++j) {
            int nd = j * 16 + r16, kb2 = rhi * 8;
            v16h b0 = ldfrag(&Vt[nd * LS + kb2],      &Vt[nd * LS + kb2 + 16]);
            v16h b1 = ldfrag(&Vt[nd * LS + 32 + kb2], &Vt[nd * LS + 32 + kb2 + 16]);
            o_acc[j] = wmma16(pa0, b0, o_acc[j]);
            o_acc[j] = wmma16(pa1, b1, o_acc[j]);
        }
    }

    // ---- normalize and write att[b, t, h*64 + d] (f16) ----
    #pragma unroll
    for (int j = 0; j < 4; ++j) {
        int d = j * 16 + r16;
        #pragma unroll
        for (int e = 0; e < 8; ++e) {
            int tg = tblk * 64 + wave * 16 + rhi * 8 + e;
            att[headoff + (size_t)tg * Eh + d] = (_Float16)(o_acc[j][e] / lrow[e]);
        }
    }
}

// ------------------------------- launcher ----------------------------------
extern "C" void kernel_launch(void* const* d_in, const int* in_sizes, int n_in,
                              void* d_out, int out_size, void* d_ws, size_t ws_size,
                              hipStream_t stream) {
    const int*   idx  = (const int*)  d_in[0];
    const float* tok  = (const float*)d_in[1];
    const float* pos  = (const float*)d_in[2];
    const float* Wq   = (const float*)d_in[3];
    const float* Wk   = (const float*)d_in[4];
    const float* Wv   = (const float*)d_in[5];
    const float* Wo   = (const float*)d_in[6];
    const float* bo   = (const float*)d_in[7];
    const float* W1   = (const float*)d_in[8];
    const float* b1   = (const float*)d_in[9];
    const float* W2   = (const float*)d_in[10];
    const float* b2   = (const float*)d_in[11];
    const float* ln1g = (const float*)d_in[12];
    const float* ln1b = (const float*)d_in[13];
    const float* ln2g = (const float*)d_in[14];
    const float* ln2b = (const float*)d_in[15];
    const float* lnfg = (const float*)d_in[16];
    const float* lnfb = (const float*)d_in[17];
    const float* Wlm  = (const float*)d_in[18];
    const float* blm  = (const float*)d_in[19];
    float* logits = (float*)d_out;

    // workspace carve
    char* w = (char*)d_ws;
    size_t off = 0;
    float*     xbuf = (float*)    (w + off); off += (size_t)Mh * Eh * 4;      // 16MB
    _Float16*  hbuf = (_Float16*) (w + off); off += (size_t)Mh * Eh * 2;      //  8MB
    _Float16*  qb   = (_Float16*) (w + off); off += (size_t)Mh * Eh * 2;
    _Float16*  kb   = (_Float16*) (w + off); off += (size_t)Mh * Eh * 2;
    _Float16*  vb   = (_Float16*) (w + off); off += (size_t)Mh * Eh * 2;
    _Float16*  attb = (_Float16*) (w + off); off += (size_t)Mh * Eh * 2;
    _Float16*  m1   = (_Float16*) (w + off); off += (size_t)Mh * 4 * Eh * 2;  // 32MB

    dim3 gEE(Eh / 128, Mh / 128);          // N=1024 GEMMs
    dim3 gE4(4 * Eh / 128, Mh / 128);      // N=4096 GEMM
    dim3 gLM(Vh / 128, Mh / 128);          // N=32000 GEMM

    embed_kernel<<<Mh, 256, 0, stream>>>(idx, tok, pos, xbuf);

    for (int l = 0; l < Lh; ++l) {
        const size_t wEE = (size_t)l * Eh * Eh;
        layernorm_kernel<<<Mh, 128, 0, stream>>>(xbuf, ln1g + l * Eh, ln1b + l * Eh, hbuf);
        gemm_wmma_kernel<true, false, false, false, false><<<gEE, 256, 0, stream>>>(
            hbuf, Wq + wEE, nullptr, nullptr, nullptr, qb, Mh, Eh, Eh);
        gemm_wmma_kernel<true, false, false, false, false><<<gEE, 256, 0, stream>>>(
            hbuf, Wk + wEE, nullptr, nullptr, nullptr, kb, Mh, Eh, Eh);
        gemm_wmma_kernel<true, false, false, false, false><<<gEE, 256, 0, stream>>>(
            hbuf, Wv + wEE, nullptr, nullptr, nullptr, vb, Mh, Eh, Eh);
        attn_kernel<<<dim3(Th / 64, Bh * Hh), 128, 0, stream>>>(qb, kb, vb, attb);
        gemm_wmma_kernel<false, true, true, false, true><<<gEE, 256, 0, stream>>>(
            attb, Wo + wEE, bo + l * Eh, xbuf, xbuf, nullptr, Mh, Eh, Eh);
        layernorm_kernel<<<Mh, 128, 0, stream>>>(xbuf, ln2g + l * Eh, ln2b + l * Eh, hbuf);
        gemm_wmma_kernel<true, false, true, true, false><<<gE4, 256, 0, stream>>>(
            hbuf, W1 + (size_t)l * Eh * 4 * Eh, b1 + (size_t)l * 4 * Eh,
            nullptr, nullptr, m1, Mh, 4 * Eh, Eh);
        gemm_wmma_kernel<false, true, true, false, true><<<gEE, 256, 0, stream>>>(
            m1, W2 + (size_t)l * 4 * Eh * Eh, b2 + l * Eh, xbuf, xbuf, nullptr,
            Mh, Eh, 4 * Eh);
    }

    layernorm_kernel<<<Mh, 128, 0, stream>>>(xbuf, lnfg, lnfb, hbuf);
    gemm_wmma_kernel<false, true, true, false, false><<<gLM, 256, 0, stream>>>(
        hbuf, Wlm, blm, nullptr, logits, nullptr, Mh, Vh, Eh);
}
